// RingSelfAttention_78228534330089
// MI455X (gfx1250) — compile-verified, matrix operand order checked by reading, not run on hardware
//
#include <hip/hip_runtime.h>
#include <hip/hip_bf16.h>

typedef __bf16 bf16_t;
typedef __attribute__((ext_vector_type(16))) __bf16 v16bf;
typedef __attribute__((ext_vector_type(8)))  float  v8f;

#define SEQL   4096
#define HID    1024
#define NHEADS 16
#define HDIM   64
#define QKVN   3072
#define SCALE  0.125f                     // 1/sqrt(64)
#define LOG2E  1.4426950408889634f

#if __has_builtin(__builtin_amdgcn_exp2f)
#define EXP2F(x) __builtin_amdgcn_exp2f(x)
#else
#define EXP2F(x) exp2f(x)
#endif

__device__ __forceinline__ v8f wmma_bf16(v16bf a, v16bf b, v8f c) {
    return __builtin_amdgcn_wmma_f32_16x16x32_bf16(
        false, a, false, b, (short)0, c, false, false);
}

// ---- 16-lane max butterfly: pure-VALU v_permlane16 if available, else shfl ----
#if __has_builtin(__builtin_amdgcn_permlane16)
__device__ __forceinline__ float permlane16_f(float v, unsigned s0, unsigned s1) {
    union { float f; unsigned u; } x; x.f = v;
    x.u = __builtin_amdgcn_permlane16(x.u, x.u, s0, s1, false, false);
    return x.f;
}
__device__ __forceinline__ float redmax16(float v) {
    v = fmaxf(v, permlane16_f(v, 0x67452301u, 0xEFCDAB89u));   // lane^1
    v = fmaxf(v, permlane16_f(v, 0x54761032u, 0xDCFE98BAu));   // lane^2
    v = fmaxf(v, permlane16_f(v, 0x32107654u, 0xBA98FEDCu));   // lane^4
    v = fmaxf(v, permlane16_f(v, 0xFEDCBA98u, 0x76543210u));   // lane^8
    return v;
}
#else
__device__ __forceinline__ float redmax16(float v) {
    v = fmaxf(v, __shfl_xor(v, 1));
    v = fmaxf(v, __shfl_xor(v, 2));
    v = fmaxf(v, __shfl_xor(v, 4));
    v = fmaxf(v, __shfl_xor(v, 8));
    return v;
}
#endif

// ---------------- f32 -> bf16 convert (native RNE cvt) ----------------
__global__ void cvt_f32_bf16(const float* __restrict__ in, bf16_t* __restrict__ out, int n) {
    int i = blockIdx.x * blockDim.x + threadIdx.x;
    if (i < n) out[i] = (bf16_t)in[i];
}

// ---------------- bf16 GEMM: C[M,N] = A[M,K] @ B[N,K]^T + bias ----------------
template <bool BF16_OUT>
__global__ __launch_bounds__(256)
void gemm_bf16(const bf16_t* __restrict__ A, const bf16_t* __restrict__ B,
               const float* __restrict__ bias, void* __restrict__ Cout,
               int M, int N, int K) {
    const int lane = threadIdx.x & 31;
    const int wave = threadIdx.x >> 5;
    const int hf   = lane >> 4;
    const int l16  = lane & 15;

    const int m0 = blockIdx.y * 128 + (wave >> 2) * 64;
    const int n0 = blockIdx.x * 256 + (wave & 3) * 64;

    v8f acc[4][4];
#pragma unroll
    for (int i = 0; i < 4; ++i)
#pragma unroll
        for (int j = 0; j < 4; ++j)
#pragma unroll
            for (int e = 0; e < 8; ++e) acc[i][j][e] = 0.f;

    const bf16_t* Ab = A + (size_t)m0 * K + (size_t)l16 * K + hf * 16;
    const bf16_t* Bb = B + (size_t)n0 * K + (size_t)l16 * K + hf * 16;

    for (int k0 = 0; k0 < K; k0 += 32) {
        v16bf a[4], b[4];
#pragma unroll
        for (int i = 0; i < 4; ++i)
            a[i] = *(const v16bf*)(Ab + (size_t)i * 16 * K + k0);
#pragma unroll
        for (int j = 0; j < 4; ++j)
            b[j] = *(const v16bf*)(Bb + (size_t)j * 16 * K + k0);
#pragma unroll
        for (int i = 0; i < 4; ++i)
#pragma unroll
            for (int j = 0; j < 4; ++j)
                acc[i][j] = wmma_bf16(a[i], b[j], acc[i][j]);
    }

#pragma unroll
    for (int i = 0; i < 4; ++i) {
#pragma unroll
        for (int j = 0; j < 4; ++j) {
            const int col = n0 + j * 16 + l16;
            const float bv = bias[col];
#pragma unroll
            for (int e = 0; e < 8; ++e) {
                const int row = m0 + i * 16 + e + hf * 8;
                const float v = acc[i][j][e] + bv;
                if constexpr (BF16_OUT)
                    ((bf16_t*)Cout)[(size_t)row * N + col] = (bf16_t)v;
                else
                    ((float*)Cout)[(size_t)row * N + col] = v;
            }
        }
    }
}

// ---------------- LDS-tiled V transpose: vt[h*64+d][s] = qkv[s][2048 + h*64 + d] ----
__global__ __launch_bounds__(256)
void transpose_v(const bf16_t* __restrict__ qkv, bf16_t* __restrict__ vt) {
    __shared__ bf16_t tile[64][66];

    const int s0  = blockIdx.x * 64;
    const int h0  = blockIdx.y * 64;
    const int tx  = threadIdx.x & 63;
    const int ty  = threadIdx.x >> 6;

#pragma unroll
    for (int r = 0; r < 64; r += 4)
        tile[r + ty][tx] = qkv[(size_t)(s0 + r + ty) * QKVN + 2 * HID + h0 + tx];
    __syncthreads();
#pragma unroll
    for (int r = 0; r < 64; r += 4)
        vt[(size_t)(h0 + r + ty) * SEQL + s0 + tx] = tile[tx][r + ty];
}

// ---------------- flash attention: 1 wave == 16 Q rows of one head ----------------
__global__ __launch_bounds__(128)
void attn_kernel(const bf16_t* __restrict__ qkv, const bf16_t* __restrict__ vt,
                 bf16_t* __restrict__ ctx) {
    __shared__ __align__(32) bf16_t plds[4][16][64];   // per-wave P tile (bf16)

    const int lane = threadIdx.x & 31;
    const int wave = threadIdx.x >> 5;
    const int hf   = lane >> 4;
    const int l16  = lane & 15;

    const int head = blockIdx.x;
    const int q0   = (blockIdx.y * 4 + wave) * 16;

    // Q fragments, pre-scaled by SCALE*log2e so scores are exp2-domain logits
    const float qscale = SCALE * LOG2E;
    const bf16_t* qb = qkv + (size_t)q0 * QKVN + head * HDIM;
    v16bf qf0 = *(const v16bf*)(qb + (size_t)l16 * QKVN + hf * 16);
    v16bf qf1 = *(const v16bf*)(qb + (size_t)l16 * QKVN + 32 + hf * 16);
#pragma unroll
    for (int e = 0; e < 16; ++e) {
        qf0[e] = (bf16_t)((float)qf0[e] * qscale);
        qf1[e] = (bf16_t)((float)qf1[e] * qscale);
    }

    // ones B-fragment for WMMA row sums
    v16bf ones;
#pragma unroll
    for (int e = 0; e < 16; ++e) ones[e] = (bf16_t)1.0f;

    v8f o[4], l_acc;
#pragma unroll
    for (int j = 0; j < 4; ++j)
#pragma unroll
        for (int e = 0; e < 8; ++e) o[j][e] = 0.f;
#pragma unroll
    for (int e = 0; e < 8; ++e) l_acc[e] = 0.f;
    float m[8];
#pragma unroll
    for (int i = 0; i < 8; ++i) m[i] = -1e30f;

    const bf16_t* kbase = qkv + HID + head * HDIM;
    const bf16_t* vbase = vt + (size_t)head * HDIM * SEQL;

    for (int kt = 0; kt < SEQL; kt += 64) {
        // ---- scores: four 16x16 tiles over a 64-key step ----
        v8f st[4];
#pragma unroll
        for (int c = 0; c < 4; ++c)
#pragma unroll
            for (int e = 0; e < 8; ++e) st[c][e] = 0.f;
#pragma unroll
        for (int c = 0; c < 4; ++c) {
            const bf16_t* kb = kbase + (size_t)(kt + c * 16 + l16) * QKVN + hf * 16;
            v16bf k0 = *(const v16bf*)kb;
            v16bf k1 = *(const v16bf*)(kb + 32);
            st[c] = wmma_bf16(qf0, k0, st[c]);
            st[c] = wmma_bf16(qf1, k1, st[c]);
        }

        // ---- online softmax (exp2 domain); stage P into LDS as bf16 ----
        float corr[8];
#pragma unroll
        for (int i = 0; i < 8; ++i) {
            const float a0 = st[0][i], a1 = st[1][i], a2 = st[2][i], a3 = st[3][i];
            float v = redmax16(fmaxf(fmaxf(a0, a1), fmaxf(a2, a3)));
            const float mn = fmaxf(m[i], v);
            corr[i] = EXP2F(m[i] - mn);
            m[i] = mn;
            const int row = i + hf * 8;
            plds[wave][row][l16]      = (bf16_t)EXP2F(a0 - mn);
            plds[wave][row][16 + l16] = (bf16_t)EXP2F(a1 - mn);
            plds[wave][row][32 + l16] = (bf16_t)EXP2F(a2 - mn);
            plds[wave][row][48 + l16] = (bf16_t)EXP2F(a3 - mn);
        }
#pragma unroll
        for (int i = 0; i < 8; ++i) l_acc[i] *= corr[i];
#pragma unroll
        for (int j = 0; j < 4; ++j)
#pragma unroll
            for (int i = 0; i < 8; ++i) o[j][i] *= corr[i];

        // ---- P as two 16x32 A-fragments ----
        v16bf pa0 = *(const v16bf*)&plds[wave][l16][hf * 16];
        v16bf pa1 = *(const v16bf*)&plds[wave][l16][32 + hf * 16];

        // row sums via matrix pipe: l += P @ ones
        l_acc = wmma_bf16(pa0, ones, l_acc);
        l_acc = wmma_bf16(pa1, ones, l_acc);

        // ---- PV: accumulate 16x64 output ----
#pragma unroll
        for (int j = 0; j < 4; ++j) {
            const bf16_t* vb = vbase + (size_t)(j * 16 + l16) * SEQL + kt + hf * 16;
            v16bf vv0 = *(const v16bf*)vb;
            v16bf vv1 = *(const v16bf*)(vb + 32);
            o[j] = wmma_bf16(pa0, vv0, o[j]);
            o[j] = wmma_bf16(pa1, vv1, o[j]);
        }
    }

    // ---- normalize and emit ctx in [s][1024] bf16 layout ----
    float inv[8];
#pragma unroll
    for (int i = 0; i < 8; ++i) inv[i] = 1.0f / l_acc[i];
#pragma unroll
    for (int j = 0; j < 4; ++j) {
#pragma unroll
        for (int i = 0; i < 8; ++i) {
            const int row = q0 + i + hf * 8;
            const int col = head * HDIM + j * 16 + l16;
            ctx[(size_t)row * HID + col] = (bf16_t)(o[j][i] * inv[i]);
        }
    }
}

extern "C" void kernel_launch(void* const* d_in, const int* in_sizes, int n_in,
                              void* d_out, int out_size, void* d_ws, size_t ws_size,
                              hipStream_t stream) {
    const float* X      = (const float*)d_in[0];   // [4096,1024]
    const float* qkv_w  = (const float*)d_in[1];   // [3072,1024]
    const float* qkv_b  = (const float*)d_in[2];   // [3072]
    const float* out_w  = (const float*)d_in[3];   // [1024,1024]
    const float* out_b  = (const float*)d_in[4];   // [1024]
    float* out = (float*)d_out;

    char* ws = (char*)d_ws;
    bf16_t* xh    = (bf16_t*)(ws);                             //  8 MB
    bf16_t* wqkvh = (bf16_t*)(ws + (size_t)8  * 1024 * 1024);  //  6 MB
    bf16_t* wouth = (bf16_t*)(ws + (size_t)14 * 1024 * 1024);  //  2 MB
    bf16_t* qkvh  = (bf16_t*)(ws + (size_t)16 * 1024 * 1024);  // 24 MB
    bf16_t* vth   = (bf16_t*)(ws + (size_t)40 * 1024 * 1024);  //  8 MB
    bf16_t* ctxh  = (bf16_t*)(ws + (size_t)48 * 1024 * 1024);  //  8 MB

    // 1) bf16 conversions
    {
        int n = SEQL * HID;
        cvt_f32_bf16<<<(n + 255) / 256, 256, 0, stream>>>(X, xh, n);
        n = QKVN * HID;
        cvt_f32_bf16<<<(n + 255) / 256, 256, 0, stream>>>(qkv_w, wqkvh, n);
        n = HID * HID;
        cvt_f32_bf16<<<(n + 255) / 256, 256, 0, stream>>>(out_w, wouth, n);
    }

    // 2) QKV = X @ Wqkv^T + b   (bf16 out)
    gemm_bf16<true><<<dim3(QKVN / 256, SEQL / 128), 256, 0, stream>>>(
        xh, wqkvh, qkv_b, (void*)qkvh, SEQL, QKVN, HID);

    // 3) V transpose (LDS-tiled, both sides coalesced)
    transpose_v<<<dim3(SEQL / 64, HID / 64), 256, 0, stream>>>(qkvh, vth);

    // 4) flash attention -> ctx (bf16)
    attn_kernel<<<dim3(NHEADS, SEQL / 64), 128, 0, stream>>>(qkvh, vth, ctxh);

    // 5) out = ctx @ Wout^T + b  (f32 out)
    gemm_bf16<false><<<dim3(HID / 256, SEQL / 128), 256, 0, stream>>>(
        ctxh, wouth, out_b, (void*)out, SEQL, HID, HID);
}
